// GAT_10084583211428
// MI455X (gfx1250) — matrix operand order repaired and checked by reading, not compiled, not run on hardware
//
#include <hip/hip_runtime.h>
#include <cstdint>
#include <cstddef>

// ---------------------------------------------------------------------------
// GAT 3-layer forward for MI455X (gfx1250, wave32).
// GEMMs use v_wmma_f32_16x16x32_bf16 (fp32 accumulate) with W pre-packed into
// a contiguous-k (k = 16*half + i) B-fragment layout so the inner loop is pure b128 loads.
// Edge softmax/scatter is the bandwidth-bound part (global f32 atomics +
// ordered-uint atomic max). LayerNorm uses wave32 shuffle reductions.
// ---------------------------------------------------------------------------

typedef __attribute__((ext_vector_type(16))) __bf16          v16bf;
typedef __attribute__((ext_vector_type(16))) unsigned short  v16us;
typedef __attribute__((ext_vector_type(8)))  unsigned short  v8us;
typedef __attribute__((ext_vector_type(8)))  float           v8f;

#define LRELU_SLOPE 0.2f
#define LN_EPS      1e-5f

// ---- ordered-uint encoding so atomicMax(uint) implements float max --------
__device__ __forceinline__ unsigned ord_enc(float f) {
  unsigned u = __float_as_uint(f);
  return (u & 0x80000000u) ? ~u : (u | 0x80000000u);
}
__device__ __forceinline__ float ord_dec(unsigned u) {
  return (u & 0x80000000u) ? __uint_as_float(u & 0x7FFFFFFFu)
                           : __uint_as_float(~u);
}
// ord_enc(-inf) = ~0xFF800000 = 0x007FFFFF
#define ORD_NEG_INF 0x007FFFFFu

__device__ __forceinline__ unsigned short f32_to_bf16(float f) {
  unsigned u = __float_as_uint(f);
  return (unsigned short)((u + 0x7FFFu + ((u >> 16) & 1u)) >> 16);
}

// ---------------------------------------------------------------------------
// f32 -> bf16 (round-to-nearest-even), flat
// ---------------------------------------------------------------------------
__global__ void k_cast_bf16(const float* __restrict__ in,
                            unsigned short* __restrict__ out, long n) {
  long i = (long)blockIdx.x * blockDim.x + threadIdx.x;
  if (i >= n) return;
  out[i] = f32_to_bf16(in[i]);
}

__global__ void k_fill_u32(unsigned* __restrict__ p, unsigned v, long n) {
  long i = (long)blockIdx.x * blockDim.x + threadIdx.x;
  if (i < n) p[i] = v;
}

// acc[n, f] = bias[f]  (folds the GATConv bias into the aggregation buffer)
__global__ void k_init_acc(float* __restrict__ acc,
                           const float* __restrict__ bias, long n, int F) {
  long i = (long)blockIdx.x * blockDim.x + threadIdx.x;
  if (i >= n) return;
  acc[i] = bias[(int)(i % F)];
}

// ---------------------------------------------------------------------------
// Pack W[K,M] (f32) into bf16 WMMA B-fragment layout:
//   Bp[((kt*TN + nt)*32 + lane)*16 + i] = bf16( W[(kt*32 + (lane>>4)*16 + i)*M
//                                                + nt*16 + (lane&15)] )
// so each lane's 16 B elements are one contiguous 32-byte chunk.
// ---------------------------------------------------------------------------
__global__ void k_pack_W(const float* __restrict__ W,
                         unsigned short* __restrict__ Bp, int K, int M) {
  long total = (long)K * M;
  long i = (long)blockIdx.x * blockDim.x + threadIdx.x;
  if (i >= total) return;
  int el   = (int)(i & 15);
  int lane = (int)((i >> 4) & 31);
  long t   = i >> 9;                 // kt*TN + nt
  int TN   = M >> 4;
  int nt   = (int)(t % TN);
  int kt   = (int)(t / TN);
  int k = kt * 32 + (lane >> 4) * 16 + el;
  int m = nt * 16 + (lane & 15);
  Bp[i] = f32_to_bf16(W[(size_t)k * M + m]);
}

// ---------------------------------------------------------------------------
// WMMA GEMM: C[rows,M] = A[rows,K](bf16) * B(packed bf16), fp32 accumulate.
// One wave computes a 16 x (16*NT) output strip; the A fragment is loaded
// once per k-step and reused across NT column tiles.
// Fragment layouts: contiguous k = 16*half + i in both A and B (wave32), not the ISA table order.
// ---------------------------------------------------------------------------
template <int K, int M, int NT>
__global__ __launch_bounds__(256)
void k_gemm_wmma_bf16(const unsigned short* __restrict__ A,
                      const unsigned short* __restrict__ Bp,
                      float* __restrict__ C, int rows) {
  constexpr int KT = K / 32;   // k-steps
  constexpr int TN = M / 16;   // total column tiles
  constexpr int NG = TN / NT;  // column-tile groups
  const int wave = blockIdx.x * (blockDim.x >> 5) + (threadIdx.x >> 5);
  const int tM   = (wave / NG) << 4;
  const int ng   = wave % NG;
  if (tM >= rows) return;                 // whole wave exits together
  const int lane = threadIdx.x & 31;
  const int half = lane >> 4;
  const int l15  = lane & 15;

  int rowA = tM + l15;
  if (rowA > rows - 1) rowA = rows - 1;   // clamp keeps EXEC full
  const unsigned short* arow = A + (size_t)rowA * K;

  v8f acc[NT];
#pragma unroll
  for (int t = 0; t < NT; ++t) acc[t] = (v8f){};

#pragma unroll
  for (int kt = 0; kt < KT; ++kt) {
    // A fragment: two contiguous 16B loads per lane
    v8us alo = *(const v8us*)(arow + kt * 32 + half * 16);
    v8us ahi = *(const v8us*)(arow + kt * 32 + half * 16 + 8);
    v16us au;
#pragma unroll
    for (int i = 0; i < 8; ++i) { au[i] = alo[i]; au[8 + i] = ahi[i]; }
    v16bf af = __builtin_bit_cast(v16bf, au);

#pragma unroll
    for (int t = 0; t < NT; ++t) {
      int nt = ng * NT + t;
      // packed B fragment: one 32-byte contiguous chunk per lane
      v16us bu = *(const v16us*)(Bp + (((size_t)(kt * TN + nt) * 32 + lane) << 4));
      acc[t] = __builtin_amdgcn_wmma_f32_16x16x32_bf16(
          false, af, false, __builtin_bit_cast(v16bf, bu),
          (short)0, acc[t], false, false);
    }
  }

#pragma unroll
  for (int t = 0; t < NT; ++t) {
    int cN = (ng * NT + t) << 4;
#pragma unroll
    for (int r = 0; r < 8; ++r) {
      int rr = tM + half * 8 + r;
      if (rr < rows) C[(size_t)rr * M + cN + l15] = acc[t][r];
    }
  }
}

// ---------------------------------------------------------------------------
// Attention scores: s_src[n,h] = sum_c h[n,h,c] * a_src[h,c]  (one wave per
// (node, head); lanes stride over C then shuffle-reduce).
// ---------------------------------------------------------------------------
__global__ __launch_bounds__(256)
void k_attn_scores(const float* __restrict__ h,
                   const float* __restrict__ a_src,
                   const float* __restrict__ a_dst,
                   float* __restrict__ ssrc, float* __restrict__ sdst,
                   int N, int H, int C) {
  int warp = blockIdx.x * (blockDim.x >> 5) + (threadIdx.x >> 5);
  int lane = threadIdx.x & 31;
  long total = (long)N * H;
  if (warp >= total) return;
  int n = warp / H, hh = warp % H;
  const float* hp = h + (size_t)n * (H * C) + hh * C;
  float ps = 0.f, pd = 0.f;
  for (int c = lane; c < C; c += 32) {
    float v = hp[c];
    ps += v * a_src[hh * C + c];
    pd += v * a_dst[hh * C + c];
  }
#pragma unroll
  for (int o = 16; o > 0; o >>= 1) {
    ps += __shfl_down(ps, o);
    pd += __shfl_down(pd, o);
  }
  if (lane == 0) { ssrc[warp] = ps; sdst[warp] = pd; }
}

// ---------------------------------------------------------------------------
// Edge pass 1: leaky-relu score, store it, atomic segment-max into emax[dst].
// Edges E0..Ep-1 are implicit self-loops (n,n).
// ---------------------------------------------------------------------------
__global__ void k_edge_score_max(const int* __restrict__ src,
                                 const int* __restrict__ dst,
                                 int E0, int Ep,
                                 const float* __restrict__ ssrc,
                                 const float* __restrict__ sdst,
                                 float* __restrict__ esc,
                                 unsigned* __restrict__ emax, int H) {
  int e = blockIdx.x * blockDim.x + threadIdx.x;
  if (e >= Ep) return;
  int s, d;
  if (e < E0) { s = src[e]; d = dst[e]; } else { s = d = e - E0; }
  for (int hh = 0; hh < H; ++hh) {
    float v = ssrc[s * H + hh] + sdst[d * H + hh];
    v = (v > 0.f) ? v : LRELU_SLOPE * v;
    esc[(size_t)e * H + hh] = v;
    atomicMax(&emax[d * H + hh], ord_enc(v));
  }
}

// Edge pass 2: exp(e - max[dst]) in place, atomic segment-sum into den[dst].
__global__ void k_edge_exp_sum(const int* __restrict__ dst,
                               int E0, int Ep,
                               float* __restrict__ esc,
                               const unsigned* __restrict__ emax,
                               float* __restrict__ den, int H) {
  int e = blockIdx.x * blockDim.x + threadIdx.x;
  if (e >= Ep) return;
  int d = (e < E0) ? dst[e] : (e - E0);
  for (int hh = 0; hh < H; ++hh) {
    float m = ord_dec(emax[d * H + hh]);
    float v = __expf(esc[(size_t)e * H + hh] - m);
    esc[(size_t)e * H + hh] = v;
    atomicAdd(&den[d * H + hh], v);
  }
}

// Edge pass 3: acc[dst] += alpha * h[src]. One wave per edge; heads outer so
// alpha is computed once per (edge, head); C is a multiple of 32 so the
// feature loop has no divergence and no integer divides.
// ---------------------------------------------------------------------------
__global__ __launch_bounds__(256)
void k_edge_scatter(const int* __restrict__ src,
                    const int* __restrict__ dst,
                    int E0, int Ep,
                    const float* __restrict__ esc,
                    const float* __restrict__ den,
                    const float* __restrict__ h,
                    float* __restrict__ acc, int H, int C) {
  int e = blockIdx.x * (blockDim.x >> 5) + (threadIdx.x >> 5);
  if (e >= Ep) return;
  int lane = threadIdx.x & 31;
  int s, d;
  if (e < E0) { s = src[e]; d = dst[e]; } else { s = d = e - E0; }
  const float* hs = h + (size_t)s * (H * C);
  float* ad = acc + (size_t)d * (H * C);
  for (int hh = 0; hh < H; ++hh) {
    float alpha = esc[(size_t)e * H + hh] / (den[d * H + hh] + 1e-16f);
    int base = hh * C;
    for (int c = lane; c < C; c += 32) {
      atomicAdd(&ad[base + c], hs[base + c] * alpha);
    }
  }
}

// ---------------------------------------------------------------------------
// LayerNorm + ReLU over F features, one wave per node; features cached in
// registers (F/32 per lane).
// ---------------------------------------------------------------------------
template <int F>
__global__ __launch_bounds__(256)
void k_ln_relu(const float* __restrict__ acc,
               const float* __restrict__ g, const float* __restrict__ be,
               float* __restrict__ out, int N) {
  constexpr int PL = F / 32;
  int n = blockIdx.x * (blockDim.x >> 5) + (threadIdx.x >> 5);
  if (n >= N) return;
  int lane = threadIdx.x & 31;
  const float* ap = acc + (size_t)n * F;
  float v[PL];
  float s = 0.f;
#pragma unroll
  for (int i = 0; i < PL; ++i) { v[i] = ap[lane + 32 * i]; s += v[i]; }
#pragma unroll
  for (int o = 16; o > 0; o >>= 1) s += __shfl_down(s, o);
  float mu = __shfl(s, 0) / (float)F;
  float vs = 0.f;
#pragma unroll
  for (int i = 0; i < PL; ++i) { float t = v[i] - mu; vs += t * t; }
#pragma unroll
  for (int o = 16; o > 0; o >>= 1) vs += __shfl_down(vs, o);
  float var = __shfl(vs, 0) / (float)F;
  float rs = rsqrtf(var + LN_EPS);
#pragma unroll
  for (int i = 0; i < PL; ++i) {
    int f = lane + 32 * i;
    float o2 = (v[i] - mu) * rs * g[f] + be[f];
    out[(size_t)n * F + f] = o2 > 0.f ? o2 : 0.f;
  }
}

// ---------------------------------------------------------------------------
// Layer-3 tail: relu(acc) + x0, atomic-accumulate into per-graph sums.
// ---------------------------------------------------------------------------
__global__ void k_relu_res_pool(const float* __restrict__ acc,
                                const float* __restrict__ x0,
                                const int* __restrict__ batch,
                                float* __restrict__ sums, long total, int F) {
  long i = (long)blockIdx.x * blockDim.x + threadIdx.x;
  if (i >= total) return;
  int n = (int)(i / F), f = (int)(i % F);
  float v = acc[i];
  v = v > 0.f ? v : 0.f;
  v += x0[i];
  atomicAdd(&sums[(size_t)batch[n] * F + f], v);
}

__global__ void k_pool_counts(const int* __restrict__ batch,
                              float* __restrict__ cnt, int N) {
  int n = blockIdx.x * blockDim.x + threadIdx.x;
  if (n < N) atomicAdd(&cnt[batch[n]], 1.0f);
}

__global__ void k_pool_final(const float* __restrict__ sums,
                             const float* __restrict__ cnt,
                             float* __restrict__ out, long total, int F) {
  long i = (long)blockIdx.x * blockDim.x + threadIdx.x;
  if (i >= total) return;
  int b = (int)(i / F);
  float c = cnt[b];
  out[i] = sums[i] / (c > 1.0f ? c : 1.0f);
}

// ---------------------------------------------------------------------------
// Host side
// ---------------------------------------------------------------------------
extern "C" void kernel_launch(void* const* d_in, const int* in_sizes, int n_in,
                              void* d_out, int out_size, void* d_ws,
                              size_t ws_size, hipStream_t stream) {
  (void)n_in; (void)ws_size;
  const int D = 384;
  const float* x   = (const float*)d_in[0];
  const int*   ei  = (const int*)d_in[1];
  const int*   bat = (const int*)d_in[2];
  const float* Wm[3]  = {(const float*)d_in[3],  (const float*)d_in[9],  (const float*)d_in[15]};
  const float* Asr[3] = {(const float*)d_in[4],  (const float*)d_in[10], (const float*)d_in[16]};
  const float* Ads[3] = {(const float*)d_in[5],  (const float*)d_in[11], (const float*)d_in[17]};
  const float* Bia[3] = {(const float*)d_in[6],  (const float*)d_in[12], (const float*)d_in[18]};
  const float* Gam[2] = {(const float*)d_in[7],  (const float*)d_in[13]};
  const float* Bet[2] = {(const float*)d_in[8],  (const float*)d_in[14]};

  const int N  = in_sizes[0] / D;      // 50000
  const int E0 = in_sizes[1] / 2;      // 500000
  const int Ep = E0 + N;               // + self loops
  const int Bg = out_size / D;         // 64
  const int Hs[3] = {4, 4, 6};
  const int Cs[3] = {96, 96, 64};
  const int HMAX  = 6;
  const int* srcI = ei;
  const int* dstI = ei + E0;

  // ---- carve workspace ----
  char* ws = (char*)d_ws;
  size_t off = 0;
  auto carve = [&](size_t bytes) -> void* {
    void* p = ws + off;
    off = (off + bytes + 255) & ~(size_t)255;
    return p;
  };
  unsigned short* xb16  = (unsigned short*)carve((size_t)N * D * 2);
  unsigned short* wpack = (unsigned short*)carve((size_t)D * D * 2);
  float*    hbuf = (float*)   carve((size_t)N * D * 4);
  float*    accb = (float*)   carve((size_t)N * D * 4);
  float*    curb = (float*)   carve((size_t)N * D * 4);
  float*    ssrc = (float*)   carve((size_t)N * HMAX * 4);
  float*    sdst = (float*)   carve((size_t)N * HMAX * 4);
  unsigned* emax = (unsigned*)carve((size_t)N * HMAX * 4);
  float*    den  = (float*)   carve((size_t)N * HMAX * 4);
  float*    esc  = (float*)   carve((size_t)Ep * HMAX * 4);
  float*    sums = (float*)   carve((size_t)Bg * D * 4);
  float*    cnts = (float*)   carve((size_t)Bg * 4);

  const int TB = 256;
  auto nblk = [](long n, int t) { return (unsigned)((n + t - 1) / t); };

  for (int l = 0; l < 3; ++l) {
    const int H = Hs[l], C = Cs[l];
    const float* xin = (l == 0) ? x : curb;

    // 1. bf16 conversion of activations; pack W into B-fragment layout
    k_cast_bf16<<<nblk((long)N * D, TB), TB, 0, stream>>>(xin, xb16, (long)N * D);
    k_pack_W<<<nblk((long)D * D, TB), TB, 0, stream>>>(Wm[l], wpack, D, D);

    // 2. h = x @ W  (WMMA bf16 -> f32), 16x64 strip per wave
    {
      long waves = (long)((N + 15) / 16) * (D / 16 / 4);
      k_gemm_wmma_bf16<384, 384, 4><<<nblk(waves * 32, TB), TB, 0, stream>>>(
          xb16, wpack, hbuf, N);
    }

    // 3. attention logits per (node, head)
    k_attn_scores<<<nblk((long)N * H * 32, TB), TB, 0, stream>>>(
        hbuf, Asr[l], Ads[l], ssrc, sdst, N, H, C);

    // 4. init segment buffers + bias-preloaded accumulator
    k_fill_u32<<<nblk((long)N * H, TB), TB, 0, stream>>>(emax, ORD_NEG_INF, (long)N * H);
    k_fill_u32<<<nblk((long)N * H, TB), TB, 0, stream>>>((unsigned*)den, 0u, (long)N * H);
    k_init_acc<<<nblk((long)N * D, TB), TB, 0, stream>>>(accb, Bia[l], (long)N * D, D);

    // 5. edge softmax + scatter
    k_edge_score_max<<<nblk(Ep, TB), TB, 0, stream>>>(srcI, dstI, E0, Ep,
                                                      ssrc, sdst, esc, emax, H);
    k_edge_exp_sum<<<nblk(Ep, TB), TB, 0, stream>>>(dstI, E0, Ep, esc, emax, den, H);
    k_edge_scatter<<<nblk((long)Ep * 32, TB), TB, 0, stream>>>(
        srcI, dstI, E0, Ep, esc, den, hbuf, accb, H, C);

    // 6. post: LN+ReLU (layers 1,2) -> curb
    if (l < 2) {
      k_ln_relu<384><<<nblk((long)N * 32, TB), TB, 0, stream>>>(
          accb, Gam[l], Bet[l], curb, N);
    }
  }

  // layer-3 tail: relu + residual + mean pool over batch
  k_fill_u32<<<nblk((long)Bg * D, TB), TB, 0, stream>>>((unsigned*)sums, 0u, (long)Bg * D);
  k_fill_u32<<<nblk(Bg, TB), TB, 0, stream>>>((unsigned*)cnts, 0u, Bg);
  k_relu_res_pool<<<nblk((long)N * D, TB), TB, 0, stream>>>(
      accb, x, bat, sums, (long)N * D, D);
  k_pool_counts<<<nblk(N, TB), TB, 0, stream>>>(bat, cnts, N);
  k_pool_final<<<nblk((long)Bg * D, TB), TB, 0, stream>>>(
      sums, cnts, (float*)d_out, (long)Bg * D, D);
}
